// GNP_19756849562137
// MI455X (gfx1250) — matrix operand order repaired and checked by reading, not compiled, not run on hardware
//
#include <hip/hip_runtime.h>
#include <hip/hip_bf16.h>
#include <hip/hip_fp16.h>

typedef __attribute__((ext_vector_type(16))) _Float16 v16h;
typedef __attribute__((ext_vector_type(8)))  _Float16 v8h;
typedef __attribute__((ext_vector_type(8)))  float    v8f;
typedef __attribute__((ext_vector_type(4)))  float    v4f;

// ---------------------------------------------------------------------------
// WMMA helpers (ISA 7.12.2 layouts, wave32)
// ---------------------------------------------------------------------------
__device__ inline v8f wmma_f16(v16h a, v16h b, v8f c) {
  return __builtin_amdgcn_wmma_f32_16x16x32_f16(
      false, a, false, b, (short)0, c, false, false);
}

// A tile (16 rows x 32 K) from 16B-aligned LDS f16, vectorized:
// per lane two contiguous 8-half chunks -> 2 x ds_load_b128.
__device__ inline v16h load_A_f16v(const _Float16* lds, int ldk, int k0, int lane) {
  int row = lane & 15;
  int sel = (lane < 16) ? 0 : 1;          // kb = sel*8
  const v8h* p = (const v8h*)(lds + row * ldk + k0);
  v8h lo = p[sel];                        // K = kb..kb+7
  v8h hi = p[2 + sel];                    // K = 16+kb..16+kb+7
  return __builtin_shufflevector(lo, hi, 0, 1, 2, 3, 4, 5, 6, 7,
                                 8, 9, 10, 11, 12, 13, 14, 15);
}

// Same but fp32 source in LDS (4 x ds_load_b128 + packed cvt).
__device__ inline v16h load_A_f32v(const float* lds, int ldk, int k0, int lane) {
  int row = lane & 15;
  int sel = (lane < 16) ? 0 : 1;
  const v4f* p = (const v4f*)(lds + row * ldk + k0);
  v4f c0 = p[sel * 2], c1 = p[sel * 2 + 1];      // K kb..kb+7
  v4f c2 = p[4 + sel * 2], c3 = p[5 + sel * 2];  // K 16+kb..16+kb+7
  v16h a;
#pragma unroll
  for (int h = 0; h < 4; ++h) {
    a[h]      = (_Float16)c0[h];
    a[4 + h]  = (_Float16)c1[h];
    a[8 + h]  = (_Float16)c2[h];
    a[12 + h] = (_Float16)c3[h];
  }
  return a;
}

// B tile: K = k0..k0+31, N = n0..n0+15 from LDS row-major [K][ldn] f16
// (strided gather; built once per wave and kept in registers).
__device__ inline v16h load_B_f16(const _Float16* ldsW, int ldn, int k0, int n0, int lane) {
  int n  = n0 + (lane & 15);
  int kb = (lane < 16) ? 0 : 8;
  v16h b;
#pragma unroll
  for (int h = 0; h < 16; ++h) b[h] = ldsW[(k0 + kb + (h < 8 ? h : h + 8)) * ldn + n];
  return b;
}

// ---------------------------------------------------------------------------
// CDNA5 async global->LDS staging (ASYNCcnt path)
// ---------------------------------------------------------------------------
__device__ inline unsigned lds_off(const void* p) {
  return (unsigned)(size_t)p;  // low 32 bits of generic LDS pointer = LDS offset
}
__device__ inline void async_ld_b128(unsigned lds_byte_off, const void* gaddr) {
  asm volatile("global_load_async_to_lds_b128 %0, %1, off"
               :: "v"(lds_byte_off), "v"(gaddr) : "memory");
}
__device__ inline void wait_async() {
  asm volatile("s_wait_asynccnt 0" ::: "memory");
}

// float atomic-max via int/uint trick (location initialized to -inf)
__device__ inline void atomicMaxFloat(float* addr, float val) {
  if (val >= 0.f) atomicMax((int*)addr, __float_as_int(val));
  else            atomicMin((unsigned int*)addr, __float_as_uint(val));
}

__device__ inline float wred(float v) {  // wave32 all-lanes sum
#pragma unroll
  for (int m = 16; m > 0; m >>= 1) v += __shfl_xor(v, m, 32);
  return v;
}

// ---------------------------------------------------------------------------
// K_cvt: pre-convert weights to f16 once.
// h16 layout: [0..8191]*6 = Wq,Wk,Wv,Wskip,laW1,lbW1 ; [49152..65535] = We
// ---------------------------------------------------------------------------
__global__ void k_cvt(const float* Wq, const float* Wk, const float* Wv,
                      const float* Ws, const float* Wa, const float* Wb,
                      const float* We, _Float16* h16) {
  int i = blockIdx.x * 256 + threadIdx.x;  // 65536 total
  if (i < 49152) {
    const float* Wt[6] = {Wq, Wk, Wv, Ws, Wa, Wb};
    h16[i] = (_Float16)Wt[i >> 13][i & 8191];
  } else {
    h16[i] = (_Float16)We[i - 49152];
  }
}

// ---------------------------------------------------------------------------
// K1: fused node GEMMs  y = x(N x 64) @ W(64 x 128) + b, 64 rows per block
// ---------------------------------------------------------------------------
__global__ __launch_bounds__(128)
void k_node_gemm(const float* __restrict__ x, const _Float16* __restrict__ h16,
                 const float* bq, const float* bk, const float* bv, const float* bs,
                 const float* ba, const float* bb,
                 float* __restrict__ wsbase, int N) {
  __shared__ __align__(16) _Float16 sW[64 * 128];  // 16 KB f16 weights
  __shared__ __align__(16) float    sX[64 * 64];   // 16 KB fp32 x tile
  int tid = threadIdx.x;
  int mat = blockIdx.y;
  const float* bt[6] = {bq, bk, bv, bs, ba, bb};
  const float* bias = bt[mat];
  float* out = wsbase + (size_t)mat * (size_t)N * 128;
  int row0 = blockIdx.x * 64;

  {  // stage f16 weights (16 KB) asynchronously
    const char* wsrc = (const char*)(h16 + (size_t)mat * 8192);
    unsigned wb = lds_off(sW);
    for (int i = tid; i < 1024; i += 128)
      async_ld_b128(wb + i * 16u, wsrc + (size_t)i * 16);
  }
  if (row0 + 64 <= N) {  // x tile is 16 KB contiguous
    const char* xsrc = (const char*)(x + (size_t)row0 * 64);
    unsigned xb = lds_off(sX);
    for (int i = tid; i < 1024; i += 128)
      async_ld_b128(xb + i * 16u, xsrc + (size_t)i * 16);
    wait_async();
  } else {
    wait_async();
    for (int i = tid; i < 64 * 64; i += 128) {
      int r = i >> 6, c = i & 63;
      sX[i] = (row0 + r < N) ? x[(size_t)(row0 + r) * 64 + c] : 0.f;
    }
  }
  __syncthreads();

  int wave = tid >> 5, lane = tid & 31;
  int mb = (lane < 16) ? 0 : 8;

  // hoist B fragments: invariant across the 4 row sub-tiles
  v16h bmat[2][2];
#pragma unroll
  for (int tnum = 0; tnum < 2; ++tnum)
#pragma unroll
    for (int ks = 0; ks < 2; ++ks)
      bmat[tnum][ks] = load_B_f16(sW, 128, ks * 32, (wave * 2 + tnum) * 16, lane);

  for (int st = 0; st < 4; ++st) {
    const float* xt = sX + st * 16 * 64;
    v16h a0 = load_A_f32v(xt, 64, 0, lane);
    v16h a1 = load_A_f32v(xt, 64, 32, lane);
    int r0 = row0 + st * 16;

    for (int tnum = 0; tnum < 2; ++tnum) {
      int n0 = (wave * 2 + tnum) * 16;
      int n  = n0 + (lane & 15);
      float bvv = bias[n];
      v8f c;
#pragma unroll
      for (int j = 0; j < 8; ++j) c[j] = bvv;
      c = wmma_f16(a0, bmat[tnum][0], c);
      c = wmma_f16(a1, bmat[tnum][1], c);

      float* po = out + (size_t)(r0 + mb) * 128 + n;
      if (r0 + 16 <= N) {
#pragma unroll
        for (int j = 0; j < 8; ++j) po[(size_t)j * 128] = c[j];
      } else {
#pragma unroll
        for (int j = 0; j < 8; ++j)
          if (r0 + mb + j < N) po[(size_t)j * 128] = c[j];
      }
    }
  }
}

// ---------------------------------------------------------------------------
// K0: init accumulators (ws is poisoned by harness)
// ---------------------------------------------------------------------------
__global__ void k_init(float* nmax, float* nden, float* attn, float* bnacc, int N) {
  int i = blockIdx.x * blockDim.x + threadIdx.x;
  int tot = N * 2;
  if (i < tot) { nmax[i] = -__builtin_inff(); nden[i] = 0.f; }
  if (i < 256) bnacc[i] = 0.f;
  size_t aTot = (size_t)N * 128;
  size_t stride = (size_t)gridDim.x * blockDim.x;
  for (size_t j = i; j < aTot; j += stride) attn[j] = 0.f;
}

// ---------------------------------------------------------------------------
// K2: batch-norm stats (coalesced partials + atomics, then finalize)
// ---------------------------------------------------------------------------
__global__ __launch_bounds__(256)
void k_bnpart(const float* __restrict__ ha, float* __restrict__ bnacc, int N) {
  int tid = threadIdx.x;
  int c = tid & 127, half = tid >> 7;
  float s = 0.f, sq = 0.f;
  for (int r = blockIdx.x * 2 + half; r < N; r += 512) {  // gridDim.x == 256
    float v = ha[(size_t)r * 128 + c];
    s += v; sq += v * v;
  }
  __shared__ float rs[256], rq[256];
  rs[tid] = s; rq[tid] = sq;
  __syncthreads();
  if (tid < 128) {
    atomicAdd(&bnacc[c], rs[tid] + rs[tid + 128]);
    atomicAdd(&bnacc[128 + c], rq[tid] + rq[tid + 128]);
  }
}
__global__ void k_bnfin(const float* __restrict__ bnacc, float* __restrict__ bnst, int N) {
  int c = threadIdx.x;  // 128
  float mean = bnacc[c] / (float)N;
  float var  = bnacc[128 + c] / (float)N - mean * mean;
  bnst[c]       = mean;
  bnst[128 + c] = rsqrtf(var + 1e-5f);
}

// ---------------------------------------------------------------------------
// K3/K5: 64 edges per block; e = edge_attr @ We via WMMA (We staged once).
//  PASS 1: logits -> ws, segment max; PASS 2: scatter (v[src]+e)*alpha
// ---------------------------------------------------------------------------
template <int PASS>
__global__ __launch_bounds__(128)
void k_edge(const float* __restrict__ lu, const float* __restrict__ t,
            const float* __restrict__ msg,
            const int* __restrict__ srcA, const int* __restrict__ dstA,
            const float* __restrict__ w_time, const float* __restrict__ b_time,
            const _Float16* __restrict__ h16We,
            const float* __restrict__ q, const float* __restrict__ k,
            const float* __restrict__ v,
            float* __restrict__ wsl, float* __restrict__ nmax,
            float* __restrict__ nden, float* __restrict__ attn, int E) {
  __shared__ __align__(16) _Float16 sWe[128 * 128];  // 32 KB
  __shared__ __align__(16) _Float16 sEA[64 * 128];   // 16 KB (4 sub-tiles)
  __shared__ int   sSrc[64], sDst[64];
  __shared__ float sPart[128];  // pass1 partial logits [64 edges][2 heads]
  __shared__ float sAl[128];    // pass2 alpha

  int tid = threadIdx.x;  // 128
  int e0  = blockIdx.x * 64;

  {  // stage f16 We (32 KB) asynchronously
    const char* wsrc = (const char*)h16We;
    unsigned wb = lds_off(sWe);
    for (int i = tid; i < 2048; i += 128)
      async_ld_b128(wb + i * 16u, wsrc + (size_t)i * 16);
    wait_async();
  }
  if (tid < 64) {
    int e = e0 + tid;
    sSrc[tid] = (e < E) ? srcA[e] : 0;
    sDst[tid] = (e < E) ? dstA[e] : 0;
  }
  sPart[tid] = 0.f;
  __syncthreads();  // sSrc/sDst + sWe visible

  if (PASS == 2) {  // alphas (tid covers 64 edges x 2 heads)
    int m = tid >> 1, h = tid & 1;
    int e = e0 + m;
    float al = 0.f;
    if (e < E)
      al = wsl[(size_t)e * 2 + h] / (nden[(size_t)sDst[m] * 2 + h] + 1e-16f);
    sAl[tid] = al;
  }

  // build edge_attr [64][128]: two threads per edge, interleaved columns
  {
    int m = tid >> 1;
    int e = e0 + m;
    _Float16* row = sEA + m * 128;
    if (e < E) {
      float rt = lu[sSrc[m]] - t[e];
#pragma unroll 4
      for (int i = 0; i < 64; ++i) {
        int c = (tid & 1) + 2 * i;
        float val = (c < 64) ? cosf(rt * w_time[c] + b_time[c])
                             : msg[(size_t)e * 64 + (c - 64)];
        row[c] = (_Float16)val;
      }
    } else {
      for (int i = 0; i < 64; ++i) row[(tid & 1) + 2 * i] = (_Float16)0.f;
    }
  }
  __syncthreads();

  int wave = tid >> 5, lane = tid & 31;
  int mb = (lane < 16) ? 0 : 8;

  // hoist B fragments: invariant across the 4 edge sub-tiles
  v16h bmat[2][4];
#pragma unroll
  for (int tnum = 0; tnum < 2; ++tnum)
#pragma unroll
    for (int s = 0; s < 4; ++s)
      bmat[tnum][s] = load_B_f16(sWe, 128, s * 32, (wave * 2 + tnum) * 16, lane);

  for (int st = 0; st < 4; ++st) {
    const _Float16* ea = sEA + st * 16 * 128;
    v16h a[4];
#pragma unroll
    for (int s = 0; s < 4; ++s) a[s] = load_A_f16v(ea, 128, s * 32, lane);
    const int* sS = sSrc + st * 16;
    const int* sD = sDst + st * 16;

    for (int tnum = 0; tnum < 2; ++tnum) {
      int n0 = (wave * 2 + tnum) * 16;
      int n  = n0 + (lane & 15);
      int head = n >> 6;
      v8f c;
#pragma unroll
      for (int j = 0; j < 8; ++j) c[j] = 0.f;
#pragma unroll
      for (int s = 0; s < 4; ++s) c = wmma_f16(a[s], bmat[tnum][s], c);

      if (PASS == 1) {
#pragma unroll
        for (int j = 0; j < 8; ++j) {
          int m = mb + j;
          float kj = k[(size_t)sS[m] * 128 + n] + c[j];
          float p  = q[(size_t)sD[m] * 128 + n] * kj;
          atomicAdd(&sPart[(st * 16 + m) * 2 + head], p);  // ds_add_f32
        }
      } else {
        if (e0 + st * 16 + 16 <= E) {
#pragma unroll
          for (int j = 0; j < 8; ++j) {
            int m = mb + j;
            float val = (v[(size_t)sS[m] * 128 + n] + c[j]) * sAl[(st * 16 + m) * 2 + head];
            atomicAdd(&attn[(size_t)sD[m] * 128 + n], val);
          }
        } else {
#pragma unroll
          for (int j = 0; j < 8; ++j) {
            int m = mb + j;
            if (e0 + st * 16 + m < E) {
              float val = (v[(size_t)sS[m] * 128 + n] + c[j]) * sAl[(st * 16 + m) * 2 + head];
              atomicAdd(&attn[(size_t)sD[m] * 128 + n], val);
            }
          }
        }
      }
    }
  }

  if (PASS == 1) {
    __syncthreads();
    int m = tid >> 1, h = tid & 1;
    int e = e0 + m;
    if (e < E) {
      float lg = sPart[tid] * 0.125f;  // / sqrt(D=64)
      wsl[(size_t)e * 2 + h] = lg;
      atomicMaxFloat(&nmax[(size_t)sDst[m] * 2 + h], lg);
    }
  }
}

// ---------------------------------------------------------------------------
// K4: a = exp(logit - max[dst]); denom[dst] += a
// ---------------------------------------------------------------------------
__global__ void k_expdenom(float* __restrict__ wsl, const float* __restrict__ nmax,
                           float* __restrict__ nden, const int* __restrict__ dstA,
                           int E) {
  int i = blockIdx.x * blockDim.x + threadIdx.x;
  if (i >= E * 2) return;
  int e = i >> 1, h = i & 1;
  int d = dstA[e];
  float a = expf(wsl[i] - nmax[(size_t)d * 2 + h]);
  wsl[i] = a;
  atomicAdd(&nden[(size_t)d * 2 + h], a);
}

// ---------------------------------------------------------------------------
// K6: epilogue, 8 rows/block, wave-per-row reductions, amortized GEMM weights
// ---------------------------------------------------------------------------
__global__ __launch_bounds__(256)
void k_final(const float* __restrict__ attn, const float* __restrict__ xr,
             const float* __restrict__ haw, const float* __restrict__ hbw,
             const float* __restrict__ Wbeta,
             const float* __restrict__ ln_g, const float* __restrict__ ln_b,
             const float* __restrict__ bnstat,
             const float* __restrict__ bn_g, const float* __restrict__ bn_b,
             const float* __restrict__ laW2, const float* __restrict__ lab2,
             const float* __restrict__ ln2_g, const float* __restrict__ ln2_b,
             const float* __restrict__ lbW2, const float* __restrict__ lbb2,
             const float* __restrict__ Wout, const float* __restrict__ bout,
             float* __restrict__ out, int N) {
  __shared__ float zpast[8][128], zha[8][128], zhb[8][128], znow[8][128];
  int tid = threadIdx.x;  // 256 = 8 waves
  int w = tid >> 5, lane = tid & 31;
  int row0 = blockIdx.x * 8;
  int row = row0 + w;

  if (row < N) {
    size_t base = (size_t)row * 128;
    float o[4], xrv[4], past[4];
    float p = 0.f;
#pragma unroll
    for (int i = 0; i < 4; ++i) {
      int c = lane + 32 * i;
      o[i]   = attn[base + c];
      xrv[i] = xr[base + c];
      p += o[i] * Wbeta[c] + xrv[i] * Wbeta[128 + c] + (o[i] - xrv[i]) * Wbeta[256 + c];
    }
    float beta = 1.f / (1.f + expf(-wred(p)));
    float s = 0.f;
#pragma unroll
    for (int i = 0; i < 4; ++i) { past[i] = beta * xrv[i] + (1.f - beta) * o[i]; s += past[i]; }
    float mu = wred(s) * (1.f / 128.f);
    float vs = 0.f;
#pragma unroll
    for (int i = 0; i < 4; ++i) { float d = past[i] - mu; vs += d * d; }
    float rstd = rsqrtf(wred(vs) * (1.f / 128.f) + 1e-5f);
#pragma unroll
    for (int i = 0; i < 4; ++i) {
      int c = lane + 32 * i;
      zpast[w][c] = (past[i] - mu) * rstd * ln_g[c] + ln_b[c];
    }
    float hbv[4]; float s2 = 0.f;
#pragma unroll
    for (int i = 0; i < 4; ++i) {
      int c = lane + 32 * i;
      float hv = haw[base + c];
      zha[w][c] = (hv - bnstat[c]) * bnstat[128 + c] * bn_g[c] + bn_b[c];
      hbv[i] = hbw[base + c];
      s2 += hbv[i];
    }
    float mu2 = wred(s2) * (1.f / 128.f);
    float v2 = 0.f;
#pragma unroll
    for (int i = 0; i < 4; ++i) { float d = hbv[i] - mu2; v2 += d * d; }
    float rstd2 = rsqrtf(wred(v2) * (1.f / 128.f) + 1e-5f);
#pragma unroll
    for (int i = 0; i < 4; ++i) {
      int c = lane + 32 * i;
      zhb[w][c] = (hbv[i] - mu2) * rstd2 * ln2_g[c] + ln2_b[c];
    }
  } else {
#pragma unroll
    for (int i = 0; i < 4; ++i) {
      int c = lane + 32 * i;
      zpast[w][c] = 0.f; zha[w][c] = 0.f; zhb[w][c] = 0.f;
    }
  }
  __syncthreads();

  {
    int c = tid & 127, rs = tid >> 7;
    float acc[4];
    if (c < 64) {
#pragma unroll
      for (int r = 0; r < 4; ++r) acc[r] = lab2[c];
      for (int i = 0; i < 128; ++i) {
        float wv = laW2[i * 64 + c];
#pragma unroll
        for (int r = 0; r < 4; ++r) acc[r] += zha[rs + 2 * r][i] * wv;
      }
    } else {
      int j = c - 64;
#pragma unroll
      for (int r = 0; r < 4; ++r) acc[r] = lbb2[j];
      for (int i = 0; i < 128; ++i) {
        float wv = lbW2[i * 64 + j];
#pragma unroll
        for (int r = 0; r < 4; ++r) acc[r] += zhb[rs + 2 * r][i] * wv;
      }
    }
#pragma unroll
    for (int r = 0; r < 4; ++r) znow[rs + 2 * r][c] = fmaxf(acc[r], 0.f);
  }
  __syncthreads();

  {
    int c = tid & 127, rs = tid >> 7;
    float fo[4];
#pragma unroll
    for (int r = 0; r < 4; ++r) fo[r] = bout[c];
    for (int i = 0; i < 128; ++i) {
      float w1 = Wout[i * 128 + c];
#pragma unroll
      for (int r = 0; r < 4; ++r) fo[r] += zpast[rs + 2 * r][i] * w1;
    }
    for (int i = 0; i < 128; ++i) {
      float w2 = Wout[(128 + i) * 128 + c];
#pragma unroll
      for (int r = 0; r < 4; ++r) fo[r] += znow[rs + 2 * r][i] * w2;
    }
#pragma unroll
    for (int r = 0; r < 4; ++r) {
      int rr = row0 + rs + 2 * r;
      if (rr < N) out[(size_t)rr * 128 + c] = fo[r];
    }
  }
}

// ---------------------------------------------------------------------------
extern "C" void kernel_launch(void* const* d_in, const int* in_sizes, int n_in,
                              void* d_out, int out_size, void* d_ws, size_t ws_size,
                              hipStream_t stream) {
  const float* x      = (const float*)d_in[0];
  const float* lu     = (const float*)d_in[1];
  const float* t      = (const float*)d_in[2];
  const float* msg    = (const float*)d_in[3];
  const int*   ei     = (const int*)d_in[4];
  const float* w_time = (const float*)d_in[5];
  const float* b_time = (const float*)d_in[6];
  const float* Wq = (const float*)d_in[7];   const float* bq = (const float*)d_in[8];
  const float* Wk = (const float*)d_in[9];   const float* bk = (const float*)d_in[10];
  const float* Wv = (const float*)d_in[11];  const float* bv = (const float*)d_in[12];
  const float* We = (const float*)d_in[13];
  const float* Wskip = (const float*)d_in[14]; const float* bskip = (const float*)d_in[15];
  const float* Wbeta = (const float*)d_in[16];
  const float* ln_g  = (const float*)d_in[17]; const float* ln_b = (const float*)d_in[18];
  const float* Wout  = (const float*)d_in[19]; const float* bout = (const float*)d_in[20];
  const float* laW1  = (const float*)d_in[21]; const float* lab1 = (const float*)d_in[22];
  const float* bn_g  = (const float*)d_in[23]; const float* bn_b = (const float*)d_in[24];
  const float* laW2  = (const float*)d_in[25]; const float* lab2 = (const float*)d_in[26];
  const float* lbW1  = (const float*)d_in[27]; const float* lbb1 = (const float*)d_in[28];
  const float* ln2_g = (const float*)d_in[29]; const float* ln2_b = (const float*)d_in[30];
  const float* lbW2  = (const float*)d_in[31]; const float* lbb2 = (const float*)d_in[32];

  int N = in_sizes[1];
  int E = in_sizes[2];
  const int* srcI = ei;
  const int* dstI = ei + E;

  float* ws = (float*)d_ws;
  size_t nod = (size_t)N * 128;
  float* q    = ws;
  float* k    = q + nod;
  float* v    = k + nod;
  float* xr   = v + nod;
  float* haw  = xr + nod;
  float* hbw  = haw + nod;
  float* attn = hbw + nod;                 // N*128
  float* wsl  = attn + nod;                // E*2
  float* nmax = wsl + (size_t)E * 2;       // N*2
  float* nden = nmax + (size_t)N * 2;      // N*2
  float* bnst = nden + (size_t)N * 2;      // 256 (mean/rstd)
  float* bnac = bnst + 256;                // 256 (sum/sumsq accum)
  _Float16* h16 = (_Float16*)(bnac + 256); // 65536 f16 weights (16B aligned)

  k_cvt<<<256, 256, 0, stream>>>(Wq, Wk, Wv, Wskip, laW1, lbW1, We, h16);
  k_init<<<(2 * N + 255) / 256, 256, 0, stream>>>(nmax, nden, attn, bnac, N);

  dim3 g1((N + 63) / 64, 6);
  k_node_gemm<<<g1, 128, 0, stream>>>(x, h16, bq, bk, bv, bskip, lab1, lbb1, q, N);

  k_bnpart<<<256, 256, 0, stream>>>(haw, bnac, N);
  k_bnfin<<<1, 128, 0, stream>>>(bnac, bnst, N);

  const _Float16* h16We = h16 + 49152;
  k_edge<1><<<(E + 63) / 64, 128, 0, stream>>>(lu, t, msg, srcI, dstI, w_time,
                                               b_time, h16We, q, k, v, wsl,
                                               nmax, nden, attn, E);
  k_expdenom<<<(2 * E + 255) / 256, 256, 0, stream>>>(wsl, nmax, nden, dstI, E);
  k_edge<2><<<(E + 63) / 64, 128, 0, stream>>>(lu, t, msg, srcI, dstI, w_time,
                                               b_time, h16We, q, k, v, wsl,
                                               nmax, nden, attn, E);
  k_final<<<(N + 7) / 8, 256, 0, stream>>>(attn, xr, haw, hbw, Wbeta, ln_g, ln_b,
                                           bnst, bn_g, bn_b, laW2, lab2, ln2_g,
                                           ln2_b, lbW2, lbb2, Wout, bout,
                                           (float*)d_out, N);
}
